// VASNet_29841432772622
// MI455X (gfx1250) — compile-verified, matrix operand order checked by reading, not run on hardware
//
#include <hip/hip_runtime.h>
#include <hip/hip_bf16.h>

#define S_DIM 4096
#define B_DIM 4
#define D_DIM 1024
#define APER  256
#define BAND  544        // padded band: 34 tiles of 16, 17 k-steps of 32
#define NKT   34
#define SCALE_C 0.03125f // 1/sqrt(1024)

typedef __attribute__((ext_vector_type(16))) __bf16 v16bf;
typedef __attribute__((ext_vector_type(8)))  __bf16 v8bf;
typedef __attribute__((ext_vector_type(8)))  float  v8f;
typedef __attribute__((ext_vector_type(2)))  float  f32x2;
typedef __attribute__((ext_vector_type(4)))  unsigned int u32x4;
typedef __attribute__((ext_vector_type(8)))  int i32x8;
typedef __attribute__((ext_vector_type(4)))  int i32x4;

typedef __attribute__((address_space(3))) const v8bf*  lds_v8bf_cp;
typedef __attribute__((address_space(3))) const f32x2* lds_f32x2_cp;

// The TDM writes LDS behind the compiler's back; a staging buffer with no
// visible store gets its loads folded to undef.  Return an opaque LDS byte
// offset (0 = start of static LDS) with the array tied as an asm input so the
// allocation stays live; reads go through explicit AS(3) pointers -> ds_load.
__device__ __forceinline__ unsigned lds_base_of(const void* p) {
  unsigned off = 0u;
  asm volatile("" : "+v"(off) : "v"(p));
  return off;
}

// ---------------------------------------------------------------------------
// ISA 16-bit A-matrix 16x32 layout (CDNA5 ISA):
//   lane = half*16 + row ;  element a[j] (j<8)  -> K = 8*half + j
//                           element a[j] (j>=8) -> K = 16 + 8*half + (j-8)
// B mirrored with col = lane&15.  C/D: VGPR r -> (M = r + 8*half, N = lane&15)
// ---------------------------------------------------------------------------

__device__ __forceinline__ int kpat(int p, int half) {
  return (p < 4) ? (8 * half + 2 * p) : (16 + 8 * half + 2 * (p - 4));
}

__device__ __forceinline__ v16bf frag_zero() {
  v16bf z;
#pragma unroll
  for (int i = 0; i < 16; ++i) z[i] = (__bf16)0.0f;
  return z;
}

// bf16 row-major source: elem(row,k) = base[row*ld + k].  Two b128 loads/lane.
__device__ __forceinline__ v16bf frag_ld_bf16(const __bf16* __restrict__ base,
                                              long ld, int k0, int lane) {
  const int half = lane >> 4;
  const int row  = lane & 15;
  const __bf16* rp = base + (long)row * ld + k0 + 8 * half;
  const v8bf lo = *(const v8bf*)(rp);
  const v8bf hi = *(const v8bf*)(rp + 16);
  v16bf a;
#pragma unroll
  for (int i = 0; i < 8; ++i) { a[i] = lo[i]; a[8 + i] = hi[i]; }
  return a;
}

// bf16 strip in LDS at byte offset `base`: two ds_load_b128 per lane.
__device__ __forceinline__ v16bf frag_ld_lds_bf16(unsigned base, int ld, int k0,
                                                  int lane) {
  const int half = lane >> 4;
  const int row  = lane & 15;
  const unsigned byt = base + (unsigned)((row * ld + k0 + 8 * half) * 2);
  const v8bf lo = *(lds_v8bf_cp)(byt);
  const v8bf hi = *(lds_v8bf_cp)(byt + 32u);
  v16bf a;
#pragma unroll
  for (int i = 0; i < 8; ++i) { a[i] = lo[i]; a[8 + i] = hi[i]; }
  return a;
}

// fp32 strip in LDS at byte offset `base`, on-the-fly cvt: ds_load_b64 x8.
__device__ __forceinline__ v16bf frag_ld_lds_f32(unsigned base, int ld, int k0,
                                                 int lane) {
  const int half = lane >> 4;
  const int row  = lane & 15;
  const unsigned rbyt = base + (unsigned)((row * ld + k0) * 4);
  v16bf a;
#pragma unroll
  for (int p = 0; p < 8; ++p) {
    const int kb = kpat(p, half);
    const f32x2 f = *(lds_f32x2_cp)(rbyt + (unsigned)(kb * 4));
    a[2 * p]     = (__bf16)f[0];
    a[2 * p + 1] = (__bf16)f[1];
  }
  return a;
}

// fp32 row-major source (generic, for the LDS alpha buffer in attn).
__device__ __forceinline__ v16bf frag_ld_f32(const float* __restrict__ base,
                                             long ld, int k0, int lane) {
  const int half = lane >> 4;
  const int row  = lane & 15;
  const float* rp = base + (long)row * ld + k0;
  v16bf a;
#pragma unroll
  for (int p = 0; p < 8; ++p) {
    const int kb = kpat(p, half);
    const float2 f = *(const float2*)(rp + kb);
    a[2 * p]     = (__bf16)f.x;
    a[2 * p + 1] = (__bf16)f.y;
  }
  return a;
}

// bf16 row-major B (elem(k,n) = base[k*ld + n]) with per-k range guard.
__device__ __forceinline__ v16bf frag_ld_rm_guard_bf16(
    const __bf16* __restrict__ base, long ld, int k0, int kglob0, int lane) {
  const int half = lane >> 4;
  const int col  = lane & 15;
  v16bf r;
#pragma unroll
  for (int p = 0; p < 8; ++p) {
    const int kb = kpat(p, half);
#pragma unroll
    for (int u = 0; u < 2; ++u) {
      const int kk = k0 + kb + u;
      const bool ok = (unsigned)(kglob0 + kk) < (unsigned)S_DIM;
      r[2 * p + u] = ok ? base[(long)kk * ld + col] : (__bf16)0.0f;
    }
  }
  return r;
}

__device__ __forceinline__ void store_tile_bf16(__bf16* __restrict__ out, long ld,
                                                v8f c, int lane) {
  const int half = lane >> 4, col = lane & 15;
#pragma unroll
  for (int r = 0; r < 8; ++r)
    out[(long)(r + 8 * half) * ld + col] = (__bf16)c[r];
}

#define WMMA_BF16(A, Bm, C) \
  __builtin_amdgcn_wmma_f32_16x16x32_bf16(false, (A), false, (Bm), (short)0, (C), false, false)

// ---------------------------------------------------------------------------
// TDM: stage a 2D strip (rows x cols, row stride in elements) into LDS.
// D# layout per CDNA5 ISA.
// ds_code: 0=1B 1=2B 2=4B 3=8B.  lds_off: byte offset in workgroup LDS.
// (6-arg builtin form: clang-23 / therock-10.0 headers lane)
// ---------------------------------------------------------------------------
__device__ __forceinline__ void tdm_load_strip(const void* gaddr, unsigned lds_off,
                                               unsigned ds_code, unsigned cols,
                                               unsigned rows, unsigned stride_elems) {
  const unsigned long long ga = (unsigned long long)gaddr;
  u32x4 g0;
  g0[0] = 1u;                                    // count=1 user descriptor
  g0[1] = lds_off;                               // lds_addr (bytes)
  g0[2] = (unsigned)(ga & 0xFFFFFFFFu);          // global_addr[31:0]
  g0[3] = (unsigned)((ga >> 32) & 0x1FFFFFFu)    // global_addr[56:32]
          | (2u << 30);                          // type = 2 ("image")
  i32x8 g1;
  g1[0] = (int)(ds_code << 16);                  // wg_mask=0 | data_size
  g1[1] = (int)((cols & 0xFFFFu) << 16);         // tensor_dim0[15:0] @ bits 63:48
  g1[2] = (int)(((cols >> 16) & 0xFFFFu) |       // tensor_dim0[31:16]
                ((rows & 0xFFFFu) << 16));       // tensor_dim1[15:0]
  g1[3] = (int)(((rows >> 16) & 0xFFFFu) |       // tensor_dim1[31:16]
                ((cols & 0xFFFFu) << 16));       // tile_dim0
  g1[4] = (int)(rows & 0xFFFFu);                 // tile_dim1 (tile_dim2 = 0)
  g1[5] = (int)stride_elems;                     // tensor_dim0_stride[31:0]
  g1[6] = 0;                                     // stride hi / dim1_stride lo
  g1[7] = 0;
  const i32x4 z4 = {0, 0, 0, 0};                 // groups 2/3 unused (2D tensor)
  const i32x8 z8 = {0, 0, 0, 0, 0, 0, 0, 0};
  __builtin_amdgcn_tensor_load_to_lds(g0, g1, z4, z4, z8, 0);
}

// ---------------------------------------------------------------------------
// Elementwise bf16 staging: x (S,B,D) -> xbh (B,S,D) bf16 ; W fp32 -> bf16
// ---------------------------------------------------------------------------
__global__ __launch_bounds__(256) void cvt_x_kernel(const float* __restrict__ x,
                                                    __bf16* __restrict__ xbh) {
  const int m = blockIdx.x;          // m = b*S + s
  const int b = m / S_DIM, s = m % S_DIM;
  const float* src = x + ((long)s * B_DIM + b) * D_DIM;
  __bf16* dst = xbh + (long)m * D_DIM;
  const int t = threadIdx.x;
#pragma unroll
  for (int i = 0; i < D_DIM / 256; ++i) dst[t + 256 * i] = (__bf16)src[t + 256 * i];
}

__global__ __launch_bounds__(256) void cvt_w_kernel(const float* __restrict__ w,
                                                    __bf16* __restrict__ wh) {
  const long base = (long)blockIdx.x * 1024 + threadIdx.x;
#pragma unroll
  for (int i = 0; i < 4; ++i) wh[base + 256 * i] = (__bf16)w[base + 256 * i];
}

// ---------------------------------------------------------------------------
// Kernel 1: K/Q/V = xb @ W^T.  8 waves/block share one TDM-staged A strip
// (16 x 1024 bf16 = 32 KB LDS); each wave owns 2 n-tiles x 3 outputs.
// ---------------------------------------------------------------------------
__global__ __launch_bounds__(256) void qkv_kernel(
    const __bf16* __restrict__ xbh,
    const __bf16* __restrict__ Wk, const __bf16* __restrict__ Wq,
    const __bf16* __restrict__ Wv,
    __bf16* __restrict__ Kb, __bf16* __restrict__ Qb, __bf16* __restrict__ Vb) {
  __shared__ __align__(16) __bf16 sA[16 * D_DIM];   // at LDS offset 0
  const int lane   = threadIdx.x & 31;
  const int wave   = threadIdx.x >> 5;
  const int tile_m = blockIdx.x;                       // over B*S/16
  const int n0     = blockIdx.y * 256 + wave * 32;     // 8 waves x 32 cols
  const __bf16* Ag = xbh + (long)tile_m * 16 * D_DIM;
  if (threadIdx.x < 32) {
    tdm_load_strip(Ag, 0u, 1u, D_DIM, 16u, D_DIM);
    __builtin_amdgcn_s_wait_tensorcnt(0);
  }
  __syncthreads();
  const unsigned sAb = lds_base_of(sA);

  const __bf16* Bk0 = Wk + (long)n0 * D_DIM;  const __bf16* Bk1 = Bk0 + 16 * D_DIM;
  const __bf16* Bq0 = Wq + (long)n0 * D_DIM;  const __bf16* Bq1 = Bq0 + 16 * D_DIM;
  const __bf16* Bv0 = Wv + (long)n0 * D_DIM;  const __bf16* Bv1 = Bv0 + 16 * D_DIM;
  v8f ck0 = {}; v8f ck1 = {}; v8f cq0 = {}; v8f cq1 = {}; v8f cv0 = {}; v8f cv1 = {};
  for (int k0 = 0; k0 < D_DIM; k0 += 32) {
    const v16bf a = frag_ld_lds_bf16(sAb, D_DIM, k0, lane);   // ds_load_b128 x2
    ck0 = WMMA_BF16(a, frag_ld_bf16(Bk0, D_DIM, k0, lane), ck0);
    ck1 = WMMA_BF16(a, frag_ld_bf16(Bk1, D_DIM, k0, lane), ck1);
    cq0 = WMMA_BF16(a, frag_ld_bf16(Bq0, D_DIM, k0, lane), cq0);
    cq1 = WMMA_BF16(a, frag_ld_bf16(Bq1, D_DIM, k0, lane), cq1);
    cv0 = WMMA_BF16(a, frag_ld_bf16(Bv0, D_DIM, k0, lane), cv0);
    cv1 = WMMA_BF16(a, frag_ld_bf16(Bv1, D_DIM, k0, lane), cv1);
  }
  const long base = (long)tile_m * 16 * D_DIM + n0;
  store_tile_bf16(Kb + base, D_DIM, ck0, lane); store_tile_bf16(Kb + base + 16, D_DIM, ck1, lane);
  store_tile_bf16(Qb + base, D_DIM, cq0, lane); store_tile_bf16(Qb + base + 16, D_DIM, cq1, lane);
  store_tile_bf16(Vb + base, D_DIM, cv0, lane); store_tile_bf16(Vb + base + 16, D_DIM, cv1, lane);
}

// ---------------------------------------------------------------------------
// Kernel 2: banded attention, all-bf16 operands.  One wave per (b, 16 queries).
// ---------------------------------------------------------------------------
__global__ __launch_bounds__(32) void attn_kernel(
    const __bf16* __restrict__ Qb, const __bf16* __restrict__ Kb,
    const __bf16* __restrict__ Vb, __bf16* __restrict__ Cb) {
  __shared__ float sc[16 * BAND];  // 34 KB of the 320 KB WGP LDS
  const int lane = threadIdx.x;
  const int half = lane >> 4, col = lane & 15;
  const int blk  = blockIdx.x;
  const int b    = blk / (S_DIM / 16);
  const int q0   = (blk % (S_DIM / 16)) * 16;
  const int keyb = q0 - APER;
  const __bf16* Aq = Qb + ((long)b * S_DIM + q0) * D_DIM;

  // ---- scores: 2 key-tiles per Q-fragment load ----
  for (int t = 0; t < NKT; t += 2) {
    const int kt0 = keyb + t * 16, kt1 = kt0 + 16;
    const bool cv0 = (unsigned)(kt0 + col) < (unsigned)S_DIM;
    const bool cv1 = (unsigned)(kt1 + col) < (unsigned)S_DIM;
    const __bf16* Bk0 = Kb + ((long)b * S_DIM + kt0) * D_DIM;
    const __bf16* Bk1 = Kb + ((long)b * S_DIM + kt1) * D_DIM;
    v8f a0 = {}; v8f a1 = {};
    for (int k0 = 0; k0 < D_DIM; k0 += 32) {
      const v16bf aq = frag_ld_bf16(Aq, D_DIM, k0, lane);
      const v16bf b0 = cv0 ? frag_ld_bf16(Bk0, D_DIM, k0, lane) : frag_zero();
      a0 = WMMA_BF16(aq, b0, a0);
      const v16bf b1 = cv1 ? frag_ld_bf16(Bk1, D_DIM, k0, lane) : frag_zero();
      a1 = WMMA_BF16(aq, b1, a1);
    }
#pragma unroll
    for (int r = 0; r < 8; ++r) {
      sc[(r + 8 * half) * BAND + t * 16 + col]        = a0[r] * SCALE_C;
      sc[(r + 8 * half) * BAND + (t + 1) * 16 + col]  = a1[r] * SCALE_C;
    }
  }
  __syncthreads();

  // ---- masked softmax, one lane per query row ----
  if (lane < 16) {
    const int q = q0 + lane;
    float* row = sc + lane * BAND;
    const float ninf = -__builtin_inff();
    float mx = ninf;
    for (int j = 0; j < BAND; ++j) {
      const int key = keyb + j;
      const bool ok = (key >= 0) && (key < S_DIM) && (key != q) &&
                      (key >= q - APER) && (key <= q + APER);
      const float e = ok ? row[j] : ninf;
      row[j] = e;
      mx = fmaxf(mx, e);
    }
    float sum = 0.0f;
    for (int j = 0; j < BAND; ++j) {
      const float e = __expf(row[j] - mx);
      row[j] = e;
      sum += e;
    }
    const float inv = 1.0f / sum;
    for (int j = 0; j < BAND; ++j) row[j] *= inv;
  }
  __syncthreads();

  // ---- C = alpha @ V : 4 n-tiles per alpha-fragment load ----
  for (int ng = 0; ng < D_DIM / 64; ++ng) {
    v8f acc0 = {}; v8f acc1 = {}; v8f acc2 = {}; v8f acc3 = {};
    for (int k0 = 0; k0 < BAND; k0 += 32) {
      const v16bf a = frag_ld_f32(sc, BAND, k0, lane);   // alpha (LDS, fp32)
      const __bf16* Bv = Vb + ((long)b * S_DIM + keyb) * D_DIM + ng * 64;
      acc0 = WMMA_BF16(a, frag_ld_rm_guard_bf16(Bv,      D_DIM, k0, keyb, lane), acc0);
      acc1 = WMMA_BF16(a, frag_ld_rm_guard_bf16(Bv + 16, D_DIM, k0, keyb, lane), acc1);
      acc2 = WMMA_BF16(a, frag_ld_rm_guard_bf16(Bv + 32, D_DIM, k0, keyb, lane), acc2);
      acc3 = WMMA_BF16(a, frag_ld_rm_guard_bf16(Bv + 48, D_DIM, k0, keyb, lane), acc3);
    }
    __bf16* outp = Cb + ((long)b * S_DIM + q0) * D_DIM + ng * 64;
    store_tile_bf16(outp,      D_DIM, acc0, lane);
    store_tile_bf16(outp + 16, D_DIM, acc1, lane);
    store_tile_bf16(outp + 32, D_DIM, acc2, lane);
    store_tile_bf16(outp + 48, D_DIM, acc3, lane);
  }
}

// ---------------------------------------------------------------------------
// Kernel 3: Y = C @ Wp^T + xb.  8 waves share TDM-staged C strip (32 KB LDS);
// each wave owns 4 n-tiles.
// ---------------------------------------------------------------------------
__global__ __launch_bounds__(256) void proj_res_kernel(
    const __bf16* __restrict__ Cb, const __bf16* __restrict__ Wp,
    const float* __restrict__ x, float* __restrict__ Y) {
  __shared__ __align__(16) __bf16 sA[16 * D_DIM];
  const int lane   = threadIdx.x & 31;
  const int wave   = threadIdx.x >> 5;
  const int tile_m = blockIdx.x;
  const int n0     = blockIdx.y * 512 + wave * 64;
  const int b  = tile_m / (S_DIM / 16);
  const int s0 = (tile_m % (S_DIM / 16)) * 16;
  if (threadIdx.x < 32) {
    tdm_load_strip(Cb + (long)tile_m * 16 * D_DIM, 0u, 1u, D_DIM, 16u, D_DIM);
    __builtin_amdgcn_s_wait_tensorcnt(0);
  }
  __syncthreads();
  const unsigned sAb = lds_base_of(sA);

  const __bf16* Bw = Wp + (long)n0 * D_DIM;
  v8f c0 = {}; v8f c1 = {}; v8f c2 = {}; v8f c3 = {};
  for (int k0 = 0; k0 < D_DIM; k0 += 32) {
    const v16bf a = frag_ld_lds_bf16(sAb, D_DIM, k0, lane);
    c0 = WMMA_BF16(a, frag_ld_bf16(Bw,              D_DIM, k0, lane), c0);
    c1 = WMMA_BF16(a, frag_ld_bf16(Bw + 16 * D_DIM, D_DIM, k0, lane), c1);
    c2 = WMMA_BF16(a, frag_ld_bf16(Bw + 32 * D_DIM, D_DIM, k0, lane), c2);
    c3 = WMMA_BF16(a, frag_ld_bf16(Bw + 48 * D_DIM, D_DIM, k0, lane), c3);
  }
  const int half = lane >> 4, col = lane & 15;
  v8f* accs[4] = {&c0, &c1, &c2, &c3};
#pragma unroll
  for (int j = 0; j < 4; ++j) {
    const int n = n0 + j * 16 + col;
#pragma unroll
    for (int r = 0; r < 8; ++r) {
      const int s = s0 + r + 8 * half;
      const float res = x[((long)s * B_DIM + b) * D_DIM + n];
      Y[((long)b * S_DIM + s) * D_DIM + n] = (*accs[j])[r] + res;
    }
  }
}

// ---------------------------------------------------------------------------
// Kernel 4: Z = relu(Y @ W1^T + b1).  TDM stages the fp32 Y strip (64 KB LDS);
// on-the-fly cvt from LDS; 8 waves x 4 n-tiles.
// ---------------------------------------------------------------------------
__global__ __launch_bounds__(256) void ffn_kernel(
    const float* __restrict__ Y, const __bf16* __restrict__ W1,
    const float* __restrict__ b1, float* __restrict__ Z) {
  __shared__ __align__(16) float sY[16 * D_DIM];  // 64 KB
  const int lane   = threadIdx.x & 31;
  const int wave   = threadIdx.x >> 5;
  const int tile_m = blockIdx.x;
  const int n0     = blockIdx.y * 512 + wave * 64;
  if (threadIdx.x < 32) {
    tdm_load_strip(Y + (long)tile_m * 16 * D_DIM, 0u, 2u, D_DIM, 16u, D_DIM);
    __builtin_amdgcn_s_wait_tensorcnt(0);
  }
  __syncthreads();
  const unsigned sYb = lds_base_of(sY);

  const __bf16* Bw = W1 + (long)n0 * D_DIM;
  v8f c0 = {}; v8f c1 = {}; v8f c2 = {}; v8f c3 = {};
  for (int k0 = 0; k0 < D_DIM; k0 += 32) {
    const v16bf a = frag_ld_lds_f32(sYb, D_DIM, k0, lane);
    c0 = WMMA_BF16(a, frag_ld_bf16(Bw,              D_DIM, k0, lane), c0);
    c1 = WMMA_BF16(a, frag_ld_bf16(Bw + 16 * D_DIM, D_DIM, k0, lane), c1);
    c2 = WMMA_BF16(a, frag_ld_bf16(Bw + 32 * D_DIM, D_DIM, k0, lane), c2);
    c3 = WMMA_BF16(a, frag_ld_bf16(Bw + 48 * D_DIM, D_DIM, k0, lane), c3);
  }
  const int half = lane >> 4, col = lane & 15;
  v8f* accs[4] = {&c0, &c1, &c2, &c3};
#pragma unroll
  for (int j = 0; j < 4; ++j) {
    const int n = n0 + j * 16 + col;
    const float bias = b1[n];
#pragma unroll
    for (int r = 0; r < 8; ++r)
      Z[(long)tile_m * 16 * D_DIM + (long)(r + 8 * half) * D_DIM + n] =
          fmaxf((*accs[j])[r] + bias, 0.0f);
  }
}

// ---------------------------------------------------------------------------
// LayerNorm (in place fp32, one 256-thread block per row)
// ---------------------------------------------------------------------------
__device__ __forceinline__ float block_sum(float v, float* sbuf) {
  const int t = threadIdx.x;
  sbuf[t] = v;
  __syncthreads();
  for (int off = 128; off > 0; off >>= 1) {
    if (t < off) sbuf[t] += sbuf[t + off];
    __syncthreads();
  }
  const float r = sbuf[0];
  __syncthreads();
  return r;
}

__global__ __launch_bounds__(256) void ln_kernel(
    float* __restrict__ Y, const float* __restrict__ gamma,
    const float* __restrict__ beta) {
  __shared__ float sbuf[256];
  float* rp = Y + (long)blockIdx.x * D_DIM;
  const int t = threadIdx.x;
  float v[4];
  float s = 0.0f;
#pragma unroll
  for (int i = 0; i < 4; ++i) { v[i] = rp[t + 256 * i]; s += v[i]; }
  s = block_sum(s, sbuf);
  const float mean = s * (1.0f / D_DIM);
  float s2 = 0.0f;
#pragma unroll
  for (int i = 0; i < 4; ++i) { const float d = v[i] - mean; s2 += d * d; }
  s2 = block_sum(s2, sbuf);
  const float inv = rsqrtf(s2 * (1.0f / D_DIM) + 1e-6f);
#pragma unroll
  for (int i = 0; i < 4; ++i) {
    const int j = t + 256 * i;
    rp[j] = (v[i] - mean) * inv * gamma[j] + beta[j];
  }
}

// ---------------------------------------------------------------------------
// Head: out[s,b] = sigmoid(dot(Z[b,s,:], W2) + b2)
// ---------------------------------------------------------------------------
__global__ __launch_bounds__(256) void head_kernel(
    const float* __restrict__ Z, const float* __restrict__ W2,
    const float* __restrict__ b2, float* __restrict__ out) {
  __shared__ float sbuf[256];
  const long m = blockIdx.x;               // m = b*S + s
  const int b = (int)(m / S_DIM), s = (int)(m % S_DIM);
  const float* rp = Z + m * D_DIM;
  const int t = threadIdx.x;
  float acc = 0.0f;
#pragma unroll
  for (int i = 0; i < D_DIM / 256; ++i)
    acc += rp[t + 256 * i] * W2[t + 256 * i];
  acc = block_sum(acc, sbuf);
  if (t == 0)
    out[(long)s * B_DIM + b] = 1.0f / (1.0f + __expf(-(acc + b2[0])));
}

// ---------------------------------------------------------------------------
extern "C" void kernel_launch(void* const* d_in, const int* in_sizes, int n_in,
                              void* d_out, int out_size, void* d_ws, size_t ws_size,
                              hipStream_t stream) {
  const float* x     = (const float*)d_in[0];
  const float* Wk    = (const float*)d_in[1];
  const float* Wq    = (const float*)d_in[2];
  const float* Wv    = (const float*)d_in[3];
  const float* Wp    = (const float*)d_in[4];
  const float* W1    = (const float*)d_in[5];
  const float* b1    = (const float*)d_in[6];
  const float* W2    = (const float*)d_in[7];
  const float* b2    = (const float*)d_in[8];
  const float* gamma = (const float*)d_in[9];
  const float* beta  = (const float*)d_in[10];
  float* out = (float*)d_out;

  const long NB = (long)B_DIM * S_DIM * D_DIM;  // 16M elements
  const long WD = (long)D_DIM * D_DIM;          // 1M elements
  char* w = (char*)d_ws;
  __bf16* xbh = (__bf16*)w;                       // 32 MB
  __bf16* Wkh = (__bf16*)(w + NB * 2);            // 5 x 2 MB weights
  __bf16* Wqh = Wkh + WD;
  __bf16* Wvh = Wqh + WD;
  __bf16* Wph = Wvh + WD;
  __bf16* W1h = Wph + WD;
  __bf16* Kb  = W1h + WD;                         // bf16 K/Q/V/C, 32 MB each
  __bf16* Qb  = Kb + NB;
  __bf16* Vb  = Qb + NB;
  __bf16* Cb  = Vb + NB;
  float*  Yf  = (float*)Kb;                       // overlays K+Q (dead after attn)
  float*  Zf  = (float*)Vb;                       // overlays V+C (dead after proj)

  // Stage bf16 operands
  cvt_x_kernel<<<B_DIM * S_DIM, 256, 0, stream>>>(x, xbh);
  cvt_w_kernel<<<WD / 1024, 256, 0, stream>>>(Wk, Wkh);
  cvt_w_kernel<<<WD / 1024, 256, 0, stream>>>(Wq, Wqh);
  cvt_w_kernel<<<WD / 1024, 256, 0, stream>>>(Wv, Wvh);
  cvt_w_kernel<<<WD / 1024, 256, 0, stream>>>(Wp, Wph);
  cvt_w_kernel<<<WD / 1024, 256, 0, stream>>>(W1, W1h);

  dim3 qkv_grid (B_DIM * S_DIM / 16, D_DIM / 256);  // (1024, 4), 8-wave blocks
  dim3 gemm_grid(B_DIM * S_DIM / 16, D_DIM / 512);  // (1024, 2), 8-wave blocks
  qkv_kernel     <<<qkv_grid,           256, 0, stream>>>(xbh, Wkh, Wqh, Wvh, Kb, Qb, Vb);
  attn_kernel    <<<B_DIM * S_DIM / 16,  32, 0, stream>>>(Qb, Kb, Vb, Cb);
  proj_res_kernel<<<gemm_grid,          256, 0, stream>>>(Cb, Wph, x, Yf);
  ln_kernel      <<<B_DIM * S_DIM,      256, 0, stream>>>(Yf, gamma, beta);
  ffn_kernel     <<<gemm_grid,          256, 0, stream>>>(Yf, W1h, b1, Zf);
  ln_kernel      <<<B_DIM * S_DIM,      256, 0, stream>>>(Zf, gamma, beta);
  head_kernel    <<<B_DIM * S_DIM,      256, 0, stream>>>(Zf, W2, b2, out);
}
